// MLAAttention_78752520340031
// MI455X (gfx1250) — matrix operand order repaired and checked by reading, not compiled, not run on hardware
//
#include <hip/hip_runtime.h>

// ---------------------------------------------------------------------------
// MLA attention for MI455X (gfx1250, wave32, WMMA bf16 16x16x32, f32 accum)
//
// Strategy: all GEMM operands are pre-converted to bf16 (and B operands
// pre-transposed to n-major [n][k]) in one memory-bound pass; everything
// (~50 MB of bf16 operands) is L2-resident (192 MB), so GEMM/attention
// fragments stream from global+LDS straight into v_wmma. LDS staging uses
// GLOBAL_LOAD_ASYNC_TO_LDS_B128 + s_wait_asynccnt (ASYNCcnt protocol).
//
// Pipeline:
//   1) bf16 casts: q->qb, kv->kvb, v_up->vupb
//   2) bf16 transposes: Wq->WqT[3072][2048], Wkv->WkvT[576][2048],
//      k_up->kupT[h][512][128]
//   3) qpb  = qb @ Wq   (bf16 out, via WqT)          [WMMA]
//   4) kvpb = kvb @ Wkv (bf16 out, via WkvT)         [WMMA]
//   5) RoPE q -> qcat[h][s][512:576] (scale folded)
//   6) pack kv -> kcat[s][576] and kcatT[512][2048] (c_kv feature-major)
//   7) q_abs = scale * q_nope @ k_up -> qcat[h][s][0:512]   [WMMA]
//   8) fused flash attention: scores = qcat@kcat' (K=576), online softmax,
//      ctx += P @ c_kv (B from kcatT), out = ctx @ v_upT    [all WMMA]
// ---------------------------------------------------------------------------

#define DEV __device__ __forceinline__

typedef __bf16 bf16_t;
typedef bf16_t v16bf __attribute__((ext_vector_type(16)));
typedef bf16_t v8bf  __attribute__((ext_vector_type(8)));
typedef float  v8f   __attribute__((ext_vector_type(8)));
typedef int    v4i_t __attribute__((ext_vector_type(4)));

DEV unsigned short f2bf(float f) {
  unsigned int u = __float_as_uint(f);
  return (unsigned short)((u + 0x7FFFu + ((u >> 16) & 1u)) >> 16);  // RNE
}
DEV float bf2f(unsigned short u) {
  return __uint_as_float((unsigned int)u << 16);
}

// ---- CDNA5 async global->LDS copy (ASYNCcnt protocol), guarded -------------
#if defined(__has_builtin)
#if __has_builtin(__builtin_amdgcn_global_load_async_to_lds_b128)
#define HAVE_ASYNC_LDS 1
#endif
#endif

DEV void cp16_g2l(const unsigned short* g, unsigned short* l) {
#ifdef HAVE_ASYNC_LDS
  __builtin_amdgcn_global_load_async_to_lds_b128(
      (__attribute__((address_space(1))) v4i_t*)g,
      (__attribute__((address_space(3))) v4i_t*)l, 0, 0);
#else
  *(uint4*)l = *(const uint4*)g;
#endif
}
DEV void wait_async_lds() {
#ifdef HAVE_ASYNC_LDS
#if __has_builtin(__builtin_amdgcn_s_wait_asynccnt)
  __builtin_amdgcn_s_wait_asynccnt(0);
#else
  asm volatile("s_wait_asynccnt 0x0" ::: "memory");
#endif
#endif
}

// ---- WMMA fragment loaders (ISA 7.12.2 layouts) ----------------------------
DEV v16bf cat8(v8bf lo, v8bf hi) {
  return __builtin_shufflevector(lo, hi, 0,1,2,3,4,5,6,7,8,9,10,11,12,13,14,15);
}
// A (16x32): lanes 0-15 row=lane, K {0..7,16..23}; lanes 16-31 row=lane-16,
// K {8..15,24..31}. Row-major source, row stride ldr elements.
DEV v16bf load_afrag(const unsigned short* base, int ldr, int lane) {
  int row = lane & 15, half = (lane >> 4) & 1;
  const unsigned short* p = base + (size_t)row * ldr + half * 8;
  v8bf lo = *(const v8bf*)p;
  v8bf hi = *(const v8bf*)(p + 16);
  return cat8(lo, hi);
}
// B (32x16) from n-major storage src[n][k]: lanes 0-15 n=lane K {0..7,16..23},
// lanes 16-31 n=lane-16 K {8..15,24..31}.
DEV v16bf load_bfrag(const unsigned short* base, int ldr, int lane) {
  int n = lane & 15, half = (lane >> 4) & 1;
  const unsigned short* p = base + (size_t)n * ldr + half * 8;
  v8bf lo = *(const v8bf*)p;
  v8bf hi = *(const v8bf*)(p + 16);
  return cat8(lo, hi);
}
DEV v8f wmma_bf16(v16bf a, v16bf b, v8f c) {
  return __builtin_amdgcn_wmma_f32_16x16x32_bf16(false, a, false, b, (short)0, c,
                                                 false, false);
}
DEV float rowmax16(float v) {
#pragma unroll
  for (int off = 1; off < 16; off <<= 1) v = fmaxf(v, __shfl_xor(v, off, 16));
  return v;
}
DEV float rowsum16(float v) {
#pragma unroll
  for (int off = 1; off < 16; off <<= 1) v += __shfl_xor(v, off, 16);
  return v;
}

// ---------------------------------------------------------------------------
// One-time bf16 conversion / transposition kernels (memory bound)
// ---------------------------------------------------------------------------
__global__ void conv_bf16(const float* __restrict__ in,
                          unsigned short* __restrict__ out) {
  int t = blockIdx.x * 256 + threadIdx.x;
  out[t] = f2bf(in[t]);
}

// out[z][n*Kd + k] = bf16(in[z][k*Nd + n])
__global__ void convT_bf16(const float* __restrict__ in,
                           unsigned short* __restrict__ out, int Kd, int Nd,
                           long inB, long outB) {
  long t = (long)blockIdx.x * 256 + threadIdx.x;
  const float* src = in + (long)blockIdx.z * inB;
  unsigned short* dst = out + (long)blockIdx.z * outB;
  int k = (int)(t / Nd), n = (int)(t % Nd);
  dst[(size_t)n * Kd + k] = f2bf(src[(size_t)k * Nd + n]);
}

// ---------------------------------------------------------------------------
// bf16 GEMM: C[M,N](bf16) = alpha * A[M,K](bf16,row-major) @ B (via BT[n][k]).
// Block = 128 thr (4 waves), tile 64x64, BK=32. A fragments straight from
// global (L2-resident); B tile async-staged to LDS once, shared by 4 waves.
// ---------------------------------------------------------------------------
__global__ __launch_bounds__(128) void gemm_bb(
    const unsigned short* __restrict__ A, const unsigned short* __restrict__ BT,
    unsigned short* __restrict__ C, int K, int lda, int ldbt, int ldc,
    long aBatch, long bBatch, long cBatch, float alpha) {
  __shared__ __align__(16) unsigned short bt_sh[64 * 40];  // [n][k], stride 40

  const int tid = threadIdx.x, lane = tid & 31, w = tid >> 5;
  const int col = lane & 15, half = (lane >> 4) & 1;
  const int m0 = blockIdx.x * 64, n0 = blockIdx.y * 64;
  A += (long)blockIdx.z * aBatch;
  BT += (long)blockIdx.z * bBatch;
  C += (long)blockIdx.z * cBatch;

  v8f acc[4] = {};
  const unsigned short* arow = A + (size_t)(m0 + 16 * w) * lda;

  for (int k0 = 0; k0 < K; k0 += 32) {
    // stage B tile (64 n-rows x 32 k) via async b128 copies
    for (int c = tid; c < 256; c += 128) {
      int n = c >> 2, ch = (c & 3) * 8;
      cp16_g2l(BT + (size_t)(n0 + n) * ldbt + k0 + ch, bt_sh + n * 40 + ch);
    }
    wait_async_lds();
    if (k0 + 32 < K)  // L2 prefetch hint for next A stripe
      __builtin_prefetch(arow + k0 + 32 + (lane << 3), 0, 1);
    __syncthreads();

    v16bf a = load_afrag(arow + k0, lda, lane);
#pragma unroll
    for (int nt = 0; nt < 4; ++nt) {
      v16bf b = load_bfrag(bt_sh + nt * 16 * 40, 40, lane);
      acc[nt] = wmma_bf16(a, b, acc[nt]);
    }
    __syncthreads();
  }

#pragma unroll
  for (int nt = 0; nt < 4; ++nt)
#pragma unroll
    for (int j = 0; j < 8; ++j) {
      size_t row = (size_t)(m0 + 16 * w + j + 8 * half);
      C[row * ldc + n0 + nt * 16 + col] = f2bf(acc[nt][j] * alpha);
    }
}

// ---------------------------------------------------------------------------
// RoPE + pack kernels (bf16 in/out)
// ---------------------------------------------------------------------------
__global__ void rope_pack_q(const unsigned short* __restrict__ qpb,
                            unsigned short* __restrict__ qcat, float scale) {
  int t = blockIdx.x * blockDim.x + threadIdx.x;  // 16*2048*32
  int i = t & 31;
  int s = (t >> 5) & 2047;
  int h = t >> 16;
  float x1 = bf2f(qpb[(size_t)s * 3072 + h * 192 + 128 + i]);
  float x2 = bf2f(qpb[(size_t)s * 3072 + h * 192 + 128 + 32 + i]);
  float inv = __powf(10000.f, -(float)i / 32.f);
  float sn, cs;
  __sincosf((float)s * inv, &sn, &cs);
  size_t base = ((size_t)h * 2048 + s) * 576 + 512;
  qcat[base + i] = f2bf((x1 * cs - x2 * sn) * scale);
  qcat[base + 32 + i] = f2bf((x1 * sn + x2 * cs) * scale);
}

__global__ void pack_k(const unsigned short* __restrict__ kvpb,
                       unsigned short* __restrict__ kcat,
                       unsigned short* __restrict__ kcatT) {
  int s = blockIdx.x, i = threadIdx.x;  // 512 threads
  unsigned short cv = kvpb[(size_t)s * 576 + i];
  kcat[(size_t)s * 576 + i] = cv;          // key-major (score B-frags)
  kcatT[(size_t)i * 2048 + s] = cv;        // feature-major (PV B-frags)
  if (i < 32) {
    float x1 = bf2f(kvpb[(size_t)s * 576 + 512 + i]);
    float x2 = bf2f(kvpb[(size_t)s * 576 + 512 + 32 + i]);
    float inv = __powf(10000.f, -(float)i / 32.f);
    float sn, cs;
    __sincosf((float)s * inv, &sn, &cs);
    kcat[(size_t)s * 576 + 512 + i] = f2bf(x1 * cs - x2 * sn);
    kcat[(size_t)s * 576 + 512 + 32 + i] = f2bf(x1 * sn + x2 * cs);
  }
}

// ---------------------------------------------------------------------------
// Fused flash attention. Block = 128 thr (4 waves), one head, 16 query rows.
// K-blocks of 64 keys: wave w scores its 16-key stripe (K=576, 18 wmma),
// cross-wave online softmax via LDS, P shared via LDS, each wave holds a
// 16x128 f32 ctx slice (8 wmma tiles, PV B-frags from L2-resident kcatT),
// then the v_up epilogue GEMM. Static LDS ~38 KB.
// ---------------------------------------------------------------------------
__global__ __launch_bounds__(128) void mla_attn(
    const unsigned short* __restrict__ qcat,
    const unsigned short* __restrict__ kcat,
    const unsigned short* __restrict__ kcatT,
    const unsigned short* __restrict__ vupb, float* __restrict__ out) {
  __shared__ __align__(16) unsigned short qs[16 * 584];     // Q rows
  __shared__ __align__(16) unsigned short psh[16 * 72];     // P tile
  __shared__ __align__(16) unsigned short ctxsh[16 * 520];  // ctx (epilogue)
  __shared__ float redmax[64], redsum[64], mrun[16], lrun[16];

  const int tid = threadIdx.x, lane = tid & 31, w = tid >> 5;
  const int col = lane & 15, half = (lane >> 4) & 1;
  const int m0 = blockIdx.x * 16;
  const int h = blockIdx.y;

  // stage Q rows (16 x 576 bf16) into LDS via async b128 copies
  const unsigned short* qsrc = qcat + ((size_t)h * 2048 + m0) * 576;
  for (int c = tid; c < 1152; c += 128) {
    int byteoff = c * 16;
    int row = byteoff / 1152;
    int rem = byteoff - row * 1152;
    cp16_g2l((const unsigned short*)((const char*)qsrc + byteoff),
             (unsigned short*)((char*)qs + row * 1168 + rem));
  }
  wait_async_lds();
  if (tid < 16) { mrun[tid] = -1e30f; lrun[tid] = 0.0f; }

  v8f cacc[8] = {};
  const int tlast = m0 + 15;

  for (int t0 = 0; t0 <= tlast; t0 += 64) {
    __syncthreads();

    // scores: S(16x16 per wave) = Q(16x576) @ K[t0+16w .. +16)^T
    v8f sacc = {};
    const unsigned short* kbase = kcat + (size_t)(t0 + 16 * w) * 576;
#pragma unroll
    for (int kk = 0; kk < 18; ++kk) {
      v16bf a = load_afrag(qs + kk * 32, 584, lane);
      v16bf b = load_bfrag(kbase + kk * 32, 576, lane);
      sacc = wmma_bf16(a, b, sacc);
    }

    // causal mask + per-wave row max
    const int tcol = t0 + 16 * w + col;
#pragma unroll
    for (int j = 0; j < 8; ++j) {
      int m = m0 + j + 8 * half;
      float s = sacc[j];
      if (tcol > m) s = -1e30f;
      sacc[j] = s;
      float v = rowmax16(s);
      if (col == 0) redmax[w * 16 + j + 8 * half] = v;
    }
    __syncthreads();

    // online softmax partial; rescale ctx accumulators; publish P (bf16)
#pragma unroll
    for (int j = 0; j < 8; ++j) {
      int r = j + 8 * half;
      float mb = fmaxf(fmaxf(redmax[r], redmax[16 + r]),
                       fmaxf(redmax[32 + r], redmax[48 + r]));
      float mold = mrun[r];
      float mnew = fmaxf(mold, mb);
      float corr = __expf(mold - mnew);
      float p = __expf(sacc[j] - mnew);
#pragma unroll
      for (int nt = 0; nt < 8; ++nt) cacc[nt][j] *= corr;
      float rs = rowsum16(p);
      if (col == 0) redsum[w * 16 + r] = rs;
      psh[r * 72 + 16 * w + col] = f2bf(p);
    }
    __syncthreads();
    if (tid < 16) {  // one lane per row updates running stats
      int r = tid;
      float mb = fmaxf(fmaxf(redmax[r], redmax[16 + r]),
                       fmaxf(redmax[32 + r], redmax[48 + r]));
      float mold = mrun[r];
      float mnew = fmaxf(mold, mb);
      float ls = redsum[r] + redsum[16 + r] + redsum[32 + r] + redsum[48 + r];
      lrun[r] = lrun[r] * __expf(mold - mnew) + ls;
      mrun[r] = mnew;
    }

    // PV: ctx(16 x 128 per wave) += P(16x64) @ c_kv slice (B from kcatT, L2)
#pragma unroll
    for (int kt = 0; kt < 2; ++kt) {
      v16bf a = load_afrag(psh + kt * 32, 72, lane);
#pragma unroll
      for (int nt = 0; nt < 8; ++nt) {
        v16bf b = load_bfrag(
            kcatT + (size_t)(128 * w + 16 * nt) * 2048 + t0 + kt * 32, 2048,
            lane);
        cacc[nt] = wmma_bf16(a, b, cacc[nt]);
      }
    }
  }

  __syncthreads();
  // normalize ctx and park in LDS: ctxsh[16][520] bf16
#pragma unroll
  for (int j = 0; j < 8; ++j) {
    int r = j + 8 * half;
    float linv = 1.0f / lrun[r];
#pragma unroll
    for (int nt = 0; nt < 8; ++nt)
      ctxsh[r * 520 + 128 * w + 16 * nt + col] = f2bf(cacc[nt][j] * linv);
  }
  __syncthreads();

  // out(16 x 32 per wave) = ctx(16x512) @ v_up[h]^T slice
  v8f oacc[2] = {};
  const unsigned short* vb = vupb + (size_t)h * 65536;
#pragma unroll
  for (int k0 = 0; k0 < 512; k0 += 32) {
    v16bf a = load_afrag(ctxsh + k0, 520, lane);
#pragma unroll
    for (int nt = 0; nt < 2; ++nt) {
      v16bf b = load_bfrag(vb + (size_t)(32 * w + 16 * nt) * 512 + k0, 512, lane);
      oacc[nt] = wmma_bf16(a, b, oacc[nt]);
    }
  }
#pragma unroll
  for (int nt = 0; nt < 2; ++nt)
#pragma unroll
    for (int j = 0; j < 8; ++j) {
      size_t row = (size_t)(m0 + j + 8 * half);
      out[row * 2048 + h * 128 + 32 * w + 16 * nt + col] = oacc[nt][j];
    }
}

// ---------------------------------------------------------------------------
// Host launcher. Workspace layout (bytes), total ~93 MB:
//   qb    @ 0        : 2048*2048*2   =  8388608
//   kvb   @ 8388608  : 2048*2048*2   =  8388608
//   vupb  @ 16777216 : 16*128*512*2  =  2097152
//   WqT   @ 18874368 : 3072*2048*2   = 12582912
//   WkvT  @ 31457280 : 576*2048*2    =  2359296
//   kupT  @ 33816576 : 16*512*128*2  =  2097152
//   qpb   @ 35913728 : 2048*3072*2   = 12582912
//   kvpb  @ 48496640 : 2048*576*2    =  2359296
//   qcat  @ 50855936 : 16*2048*576*2 = 37748736
//   kcat  @ 88604672 : 2048*576*2    =  2359296
//   kcatT @ 90963968 : 512*2048*2    =  2097152
// ---------------------------------------------------------------------------
extern "C" void kernel_launch(void* const* d_in, const int* in_sizes, int n_in,
                              void* d_out, int out_size, void* d_ws,
                              size_t ws_size, hipStream_t stream) {
  (void)in_sizes; (void)n_in; (void)out_size; (void)ws_size;
  const float* q = (const float*)d_in[0];
  const float* kv = (const float*)d_in[1];
  const float* Wq = (const float*)d_in[2];
  const float* Wkv = (const float*)d_in[3];
  const float* k_up = (const float*)d_in[4];
  const float* v_up = (const float*)d_in[5];
  float* out = (float*)d_out;

  char* ws = (char*)d_ws;
  unsigned short* qb = (unsigned short*)(ws + 0);
  unsigned short* kvb = (unsigned short*)(ws + 8388608);
  unsigned short* vupb = (unsigned short*)(ws + 16777216);
  unsigned short* WqT = (unsigned short*)(ws + 18874368);
  unsigned short* WkvT = (unsigned short*)(ws + 31457280);
  unsigned short* kupT = (unsigned short*)(ws + 33816576);
  unsigned short* qpb = (unsigned short*)(ws + 35913728);
  unsigned short* kvpb = (unsigned short*)(ws + 48496640);
  unsigned short* qcat = (unsigned short*)(ws + 50855936);
  unsigned short* kcat = (unsigned short*)(ws + 88604672);
  unsigned short* kcatT = (unsigned short*)(ws + 90963968);

  const float scale = 0.07216878364870322f;  // 1/sqrt(128+64)

  // 1) one-time bf16 casts
  conv_bf16<<<16384, 256, 0, stream>>>(q, qb);
  conv_bf16<<<16384, 256, 0, stream>>>(kv, kvb);
  conv_bf16<<<4096, 256, 0, stream>>>(v_up, vupb);
  // 2) one-time bf16 transposes (B operands to n-major [n][k])
  convT_bf16<<<24576, 256, 0, stream>>>(Wq, WqT, 2048, 3072, 0, 0);
  convT_bf16<<<4608, 256, 0, stream>>>(Wkv, WkvT, 2048, 576, 0, 0);
  convT_bf16<<<dim3(256, 1, 16), 256, 0, stream>>>(k_up, kupT, 128, 512, 65536,
                                                   65536);
  // 3) qpb = qb @ Wq
  gemm_bb<<<dim3(32, 48, 1), 128, 0, stream>>>(qb, WqT, qpb, 2048, 2048, 2048,
                                               3072, 0, 0, 0, 1.0f);
  // 4) kvpb = kvb @ Wkv
  gemm_bb<<<dim3(32, 9, 1), 128, 0, stream>>>(kvb, WkvT, kvpb, 2048, 2048, 2048,
                                              576, 0, 0, 0, 1.0f);
  // 5) roped q -> qcat[..., 512:576] (scale folded)
  rope_pack_q<<<4096, 256, 0, stream>>>(qpb, qcat, scale);
  // 6) c_kv + roped k -> kcat / kcatT
  pack_k<<<2048, 512, 0, stream>>>(kvpb, kcat, kcatT);
  // 7) q_abs = scale * q_nope @ k_up -> qcat[..., 0:512], per head via grid.z
  gemm_bb<<<dim3(32, 8, 16), 128, 0, stream>>>(qpb, kupT, qcat, 128, 3072, 128,
                                               576, 192, 65536, 1179648, scale);
  // 8) fused flash attention + v_up epilogue
  mla_attn<<<dim3(128, 16), 128, 0, stream>>>(qcat, kcat, kcatT, vupb, out);
}
